// LocationSensitiveAttention_62079457296565
// MI455X (gfx1250) — matrix operand order repaired and checked by reading, not compiled, not run on hardware
//
#include <hip/hip_runtime.h>
#include <hip/hip_bf16.h>
#include <math.h>

// ---------------------------------------------------------------------------
// LocationSensitiveAttention on MI455X (gfx1250), wave32 + WMMA bf16.
//
//   e[b,t] = wv . tanh( memory[b,t,:]@Wm + q[b]@Wq + conv(prev,cum)[b,:,t]@Wloc
//                       + (bq+bm+bloc) ) + bv        -> softmax over t -> a
//   ctx[b] = a[b,:] @ memory[b]
//
// Roofline: dominant GEMM (68.7 GFLOP bf16-WMMA) vs ~540 MB HBM -> memory
// bound. v2 change: 8-wave workgroups stream the same Wm chunk through the
// shared WGP$ in lockstep (L2 B-traffic 4.3 GB -> 0.54 GB), and tanh uses the
// gfx1250 V_TANH_F32 TRANS op (co-executes with WMMA).
// ---------------------------------------------------------------------------

#define Bn 32
#define Tn 4096
#define Dn 512
#define Cn 32
#define Kn 31

typedef __attribute__((ext_vector_type(16))) __bf16 v16bf;
typedef __attribute__((ext_vector_type(8)))  __bf16 v8bf;
typedef __attribute__((ext_vector_type(8)))  float  v8f;
typedef __attribute__((ext_vector_type(4)))  float  v4f;

#if __has_builtin(__builtin_amdgcn_tanhf)
__device__ __forceinline__ float fast_tanh(float x) { return __builtin_amdgcn_tanhf(x); }
#else
__device__ __forceinline__ float fast_tanh(float x) { return tanhf(x); }
#endif

// A-fragment (16x32 bf16, ISA 7.12.2): lane holds two 8-wide K runs at p and p+16
__device__ __forceinline__ v16bf frag_a_from_f32(const float* __restrict__ p) {
  v4f x0 = *(const v4f*)(p);
  v4f x1 = *(const v4f*)(p + 4);
  v4f y0 = *(const v4f*)(p + 16);
  v4f y1 = *(const v4f*)(p + 20);
  v16bf r;
#pragma unroll
  for (int i = 0; i < 4; ++i) {
    r[i]      = (__bf16)x0[i];
    r[4 + i]  = (__bf16)x1[i];
    r[8 + i]  = (__bf16)y0[i];
    r[12 + i] = (__bf16)y1[i];
  }
  return r;
}

// B-fragment (32x16 bf16): lane holds two 8-wide K runs at p and p+16
__device__ __forceinline__ v16bf frag_b_bf16(const __bf16* __restrict__ p) {
  union { v16bf v; v8bf h[2]; } u;
  u.h[0] = *(const v8bf*)(p);
  u.h[1] = *(const v8bf*)(p + 16);
  return u.v;
}

// --------------------------- prep kernels ----------------------------------

__global__ __launch_bounds__(256) void prep_wm_kernel(const float* __restrict__ Wm,
                                                      __bf16* __restrict__ WmT) {
  int idx = blockIdx.x * 256 + threadIdx.x;   // n*512 + k
  int n = idx >> 9, k = idx & 511;
  WmT[idx] = (__bf16)Wm[(size_t)k * Dn + n];
}

__global__ __launch_bounds__(256) void prep_wloc_kernel(const float* __restrict__ Wloc,
                                                        __bf16* __restrict__ WlocT) {
  int idx = blockIdx.x * 256 + threadIdx.x;   // d*32 + c
  int d = idx >> 5, c = idx & 31;
  WlocT[idx] = (__bf16)Wloc[(size_t)c * Dn + d];
}

__global__ __launch_bounds__(256) void qbias_kernel(const float* __restrict__ query,
                                                    const float* __restrict__ Wq,
                                                    const float* __restrict__ bq,
                                                    const float* __restrict__ bm,
                                                    const float* __restrict__ bloc,
                                                    float* __restrict__ qbias) {
  int idx = blockIdx.x * 256 + threadIdx.x;   // b*512 + d
  int b = idx >> 9, d = idx & 511;
  float s = bq[d] + bm[d] + bloc[d];
  const float* qr = query + (size_t)b * Dn;
  for (int k = 0; k < Dn; ++k) s += qr[k] * Wq[(size_t)k * Dn + d];
  qbias[idx] = s;
}

// conv1d SAME over stacked (prev, cum): convT[b,t,c]
__global__ __launch_bounds__(256) void conv_kernel(const float* __restrict__ prev,
                                                   const float* __restrict__ cum,
                                                   const float* __restrict__ Wconv,
                                                   const float* __restrict__ bconv,
                                                   float* __restrict__ convT) {
  int idx = blockIdx.x * 256 + threadIdx.x;   // (b*T + t)*32 + c
  int c = idx & 31;
  int t = (idx >> 5) & (Tn - 1);
  int b = idx >> 17;
  const float* w0 = Wconv + (size_t)(c * 2 + 0) * Kn;
  const float* w1 = Wconv + (size_t)(c * 2 + 1) * Kn;
  const float* p  = prev + (size_t)b * Tn;
  const float* q  = cum  + (size_t)b * Tn;
  float s = bconv[c];
#pragma unroll
  for (int kk = 0; kk < Kn; ++kk) {
    int tt = t + kk - (Kn / 2);
    if (tt >= 0 && tt < Tn) s += w0[kk] * p[tt] + w1[kk] * q[tt];
  }
  convT[idx] = s;
}

// --------------------------- energy (WMMA) ---------------------------------
// 8 waves per workgroup; wave w owns rows [t0+16w, t0+16w+16). All waves walk
// the same 64-column Wm chunk per iteration (barrier-synced) so the chunk
// streams through the shared WGP$ once per workgroup instead of once per wave.

__global__ __launch_bounds__(256) void energy_kernel(const float* __restrict__ memory,
                                                     const float* __restrict__ convT,
                                                     const __bf16* __restrict__ WmT,
                                                     const __bf16* __restrict__ WlocT,
                                                     const float* __restrict__ qbias,
                                                     const float* __restrict__ wv,
                                                     const float* __restrict__ bv,
                                                     const unsigned char* __restrict__ mask,
                                                     float* __restrict__ e) {
  const int wave = threadIdx.x >> 5;
  const int tile = blockIdx.x * 8 + wave;    // 16-row tile index
  const int b    = tile >> 8;                // 256 tiles per batch
  const int t0   = (tile & 255) << 4;
  const int lane = threadIdx.x & 31;
  const int half = lane >> 4;
  const int r    = lane & 15;

  // A fragments: 16 rows of memory, full K=512 (16 steps of 32), bf16 in-register
  const float* arow = memory + (size_t)(b * Tn + t0 + r) * Dn;
  v16bf afrag[16];
#pragma unroll
  for (int ks = 0; ks < 16; ++ks)
    afrag[ks] = frag_a_from_f32(arow + ks * 32 + half * 8);

  // conv-feature fragment (K = C = 32, one WMMA step)
  const float* crow = convT + (size_t)(b * Tn + t0 + r) * Cn;
  v16bf cfrag = frag_a_from_f32(crow + half * 8);

  float eacc[8];
#pragma unroll
  for (int v = 0; v < 8; ++v) eacc[v] = 0.0f;

  for (int c = 0; c < 8; ++c) {              // 8 chunks of 64 output columns
    __syncthreads();                         // keep 8 waves on the same B chunk
    const int n0 = c << 6;
    v8f acc[4] = {v8f{}, v8f{}, v8f{}, v8f{}};

    // location-feature contribution: convT(16x32) @ WlocT^T
#pragma unroll
    for (int j = 0; j < 4; ++j) {
      const __bf16* wl = WlocT + (size_t)(n0 + j * 16 + r) * Cn + half * 8;
      acc[j] = __builtin_amdgcn_wmma_f32_16x16x32_bf16(
          false, cfrag, false, frag_b_bf16(wl), (short)0, acc[j], false, false);
    }
    // main GEMM: memory(16x512) @ Wm chunk(512x64)
#pragma unroll
    for (int ks = 0; ks < 16; ++ks) {
#pragma unroll
      for (int j = 0; j < 4; ++j) {
        const __bf16* wp = WmT + (size_t)(n0 + j * 16 + r) * Dn + ks * 32 + half * 8;
        acc[j] = __builtin_amdgcn_wmma_f32_16x16x32_bf16(
            false, afrag[ks], false, frag_b_bf16(wp), (short)0, acc[j], false, false);
      }
    }
    // epilogue: + qbias, tanh, scale by wv, accumulate per-row energy partials
#pragma unroll
    for (int j = 0; j < 4; ++j) {
      const int n = n0 + j * 16 + r;         // this lane's output column
      const float qb = qbias[b * Dn + n];
      const float w  = wv[n];
#pragma unroll
      for (int v = 0; v < 8; ++v)            // row m = v + 8*half
        eacc[v] += w * fast_tanh(acc[j][v] + qb);
    }
  }

  // reduce across the 16 lanes of each half (columns) -> per-row energies
#pragma unroll
  for (int m = 1; m < 16; m <<= 1) {
#pragma unroll
    for (int v = 0; v < 8; ++v)
      eacc[v] += __shfl_xor(eacc[v], m, 32);
  }
  if (r == 0) {                              // lane 0 -> rows 0..7, lane 16 -> 8..15
    const float bvs = bv[0];
#pragma unroll
    for (int v = 0; v < 8; ++v) {
      const int t = t0 + half * 8 + v;
      float val = eacc[v] + bvs;
      if (mask[(size_t)b * Tn + t]) val = -1.0e9f;
      e[(size_t)b * Tn + t] = val;
    }
  }
}

// --------------------------- softmax + context -----------------------------

__global__ __launch_bounds__(256) void softmax_kernel(const float* __restrict__ e,
                                                      const unsigned char* __restrict__ mask,
                                                      float* __restrict__ a) {
  __shared__ float redf[256];
  __shared__ int   redi[256];
  const int b = blockIdx.x, tid = threadIdx.x;
  const float* er = e + (size_t)b * Tn;
  const unsigned char* mr = mask + (size_t)b * Tn;
  float* ar = a + (size_t)b * Tn;

  int unm = 0;
  for (int t = tid; t < Tn; t += 256) unm += mr[t] ? 0 : 1;
  redi[tid] = unm;
  __syncthreads();
  for (int s = 128; s > 0; s >>= 1) {
    if (tid < s) redi[tid] += redi[tid + s];
    __syncthreads();
  }
  const bool allm = (redi[0] == 0);          // fully-masked row: e[0] forced to 0

  float mx = -3.4e38f;
  for (int t = tid; t < Tn; t += 256) {
    float x = er[t];
    if (allm && t == 0) x = 0.0f;
    mx = fmaxf(mx, x);
  }
  redf[tid] = mx;
  __syncthreads();
  for (int s = 128; s > 0; s >>= 1) {
    if (tid < s) redf[tid] = fmaxf(redf[tid], redf[tid + s]);
    __syncthreads();
  }
  mx = redf[0];
  __syncthreads();

  float sum = 0.0f;
  for (int t = tid; t < Tn; t += 256) {
    float x = er[t];
    if (allm && t == 0) x = 0.0f;
    sum += __expf(x - mx);
  }
  redf[tid] = sum;
  __syncthreads();
  for (int s = 128; s > 0; s >>= 1) {
    if (tid < s) redf[tid] += redf[tid + s];
    __syncthreads();
  }
  const float inv = 1.0f / redf[0];
  for (int t = tid; t < Tn; t += 256) {
    float x = er[t];
    if (allm && t == 0) x = 0.0f;
    ar[t] = __expf(x - mx) * inv;
  }
}

__global__ __launch_bounds__(128) void ctx_kernel(const float* __restrict__ a,
                                                  const float* __restrict__ memory,
                                                  float* __restrict__ ctx) {
  const int b = blockIdx.x >> 2;
  const int d = ((blockIdx.x & 3) << 7) + threadIdx.x;
  const float* mb = memory + (size_t)b * Tn * Dn + d;
  const float* ab = a + (size_t)b * Tn;
  float a0 = 0.f, a1 = 0.f, a2 = 0.f, a3 = 0.f;
  for (int t = 0; t < Tn; t += 4) {
    a0 += ab[t]     * mb[(size_t)(t)     * Dn];
    a1 += ab[t + 1] * mb[(size_t)(t + 1) * Dn];
    a2 += ab[t + 2] * mb[(size_t)(t + 2) * Dn];
    a3 += ab[t + 3] * mb[(size_t)(t + 3) * Dn];
  }
  ctx[(size_t)b * Dn + d] = (a0 + a1) + (a2 + a3);
}

// --------------------------- launcher --------------------------------------

extern "C" void kernel_launch(void* const* d_in, const int* in_sizes, int n_in,
                              void* d_out, int out_size, void* d_ws, size_t ws_size,
                              hipStream_t stream) {
  (void)in_sizes; (void)n_in; (void)out_size; (void)ws_size;

  const float* query = (const float*)d_in[0];
  const float* memory = (const float*)d_in[1];
  const float* prev  = (const float*)d_in[2];
  const float* cum   = (const float*)d_in[3];
  const unsigned char* mask = (const unsigned char*)d_in[4];
  const float* Wq    = (const float*)d_in[5];
  const float* bq    = (const float*)d_in[6];
  const float* Wm    = (const float*)d_in[7];
  const float* bm    = (const float*)d_in[8];
  const float* Wconv = (const float*)d_in[9];
  const float* bconv = (const float*)d_in[10];
  const float* Wloc  = (const float*)d_in[11];
  const float* bloc  = (const float*)d_in[12];
  const float* wv    = (const float*)d_in[13];
  const float* bv    = (const float*)d_in[14];

  // workspace carve-up
  char* ws = (char*)d_ws;
  __bf16* WmT   = (__bf16*)ws;  ws += (size_t)Dn * Dn * 2;        // 512 KB
  __bf16* WlocT = (__bf16*)ws;  ws += (size_t)Dn * Cn * 2;        // 32 KB
  float*  qbias = (float*)ws;   ws += (size_t)Bn * Dn * 4;        // 64 KB
  float*  convT = (float*)ws;   ws += (size_t)Bn * Tn * Cn * 4;   // 16 MB
  float*  earr  = (float*)ws;   ws += (size_t)Bn * Tn * 4;        // 512 KB

  float* ctx_out = (float*)d_out;            // [B, D]
  float* a_out   = (float*)d_out + Bn * Dn;  // [B, T]

  prep_wm_kernel  <<<(Dn * Dn) / 256, 256, 0, stream>>>(Wm, WmT);
  prep_wloc_kernel<<<(Dn * Cn) / 256, 256, 0, stream>>>(Wloc, WlocT);
  qbias_kernel    <<<(Bn * Dn) / 256, 256, 0, stream>>>(query, Wq, bq, bm, bloc, qbias);
  conv_kernel     <<<(Bn * Tn * Cn) / 256, 256, 0, stream>>>(prev, cum, Wconv, bconv, convT);
  energy_kernel   <<<Bn * (Tn / 16) / 8, 256, 0, stream>>>(memory, convT, WmT, WlocT,
                                                           qbias, wv, bv, mask, earr);
  softmax_kernel  <<<Bn, 256, 0, stream>>>(earr, mask, a_out);
  ctx_kernel      <<<Bn * (Dn / 128), 128, 0, stream>>>(a_out, memory, ctx_out);
}
